// MCxM_GNN_16003048145031
// MI455X (gfx1250) — matrix operand order repaired and checked by reading, not compiled, not run on hardware
//
#include <hip/hip_runtime.h>
#include <hip/hip_bf16.h>

#define NN 100000
#define EE 3200000
#define HH 128
#define GG 64
#define LL 3
#define BN_EPS 1e-5f

typedef _Float16 v16h __attribute__((ext_vector_type(16)));
typedef float    v8f  __attribute__((ext_vector_type(8)));

// ---------------------------------------------------------------------------
// Degree / normalization
// ---------------------------------------------------------------------------
__global__ void deg_init_kernel(float* __restrict__ deg) {
    int i = blockIdx.x * blockDim.x + threadIdx.x;
    if (i < NN) deg[i] = 1.0f;  // self-loop contributes 1
}

__global__ void deg_accum_kernel(const int* __restrict__ ei, float* __restrict__ deg) {
    int e = blockIdx.x * blockDim.x + threadIdx.x;
    if (e < EE) atomicAdd(&deg[ei[EE + e]], 1.0f);   // cols = edge_index[1]
}

__global__ void deg_rsqrt_kernel(float* __restrict__ deg) {
    int i = blockIdx.x * blockDim.x + threadIdx.x;
    if (i < NN) {
        float d = deg[i];
        deg[i] = (d > 0.0f) ? rsqrtf(d) : 0.0f;
    }
}

// ---------------------------------------------------------------------------
// GEMM: hw = (h * mask) @ W   via v_wmma_f32_16x16x32_f16
// Block = 256 thr = 8 waves; block -> 16 rows, wave w -> cols [16w,16w+16)
// ---------------------------------------------------------------------------
__global__ __launch_bounds__(256) void gemm_mask_wmma_kernel(
    const float* __restrict__ hin, const float* __restrict__ mask,
    const float* __restrict__ Wl, float* __restrict__ hw)
{
    const int lane = threadIdx.x & 31;
    const int wave = threadIdx.x >> 5;
    const int hl   = lane >> 4;     // half of wave: 0 or 1
    const int r    = lane & 15;
    const int m0   = blockIdx.x * 16;
    const int n0   = wave * 16;
    const int row  = m0 + r;

    const float mk = mask[row];
    const float* __restrict__ hrow = hin + (size_t)row * HH;

    v8f c = {};
#pragma unroll
    for (int kc = 0; kc < 4; ++kc) {
        const int k0 = kc * 32;
        v16h a, bfr;
#pragma unroll
        for (int j = 0; j < 16; ++j) {
            // A 16x32 f16 layout: lanes 0-15 row M=r; elems 0..7 -> K = hl*8+j,
            // elems 8..15 -> K = 16 + hl*8 + (j-8)
            int ka = k0 + ((j < 8) ? (hl * 8 + j) : (16 + hl * 8 + (j - 8)));
            a[j] = (_Float16)(hrow[ka] * mk);
            // B 32x16 f16 layout: same K order as A (elems 0..7 -> K = hl*8+j, 8..15 -> K = 16 + hl*8 + (j-8)); N = r
            int kb = k0 + ((j < 8) ? (hl * 8 + j) : (16 + hl * 8 + (j - 8)));
            bfr[j] = (_Float16)Wl[(size_t)kb * HH + n0 + r];
        }
        c = __builtin_amdgcn_wmma_f32_16x16x32_f16(
            /*neg_a=*/false, a, /*neg_b=*/false, bfr,
            /*c_mod=*/(short)0, c, /*reuse_a=*/false, /*reuse_b=*/false);
    }
    // C/D 16x16 f32 layout: VGPR v -> M = v (lanes 0-15), M = v+8 (lanes 16-31)
#pragma unroll
    for (int v = 0; v < 8; ++v) {
        hw[(size_t)(m0 + v + hl * 8) * HH + n0 + r] = c[v];
    }
}

// ---------------------------------------------------------------------------
// agg = b  (bias broadcast init)
// ---------------------------------------------------------------------------
__global__ void init_agg_kernel(const float* __restrict__ bl, float* __restrict__ agg) {
    long long i = (long long)blockIdx.x * blockDim.x + threadIdx.x;
    if (i < (long long)NN * HH) agg[i] = bl[i & (HH - 1)];
}

// ---------------------------------------------------------------------------
// Edge scatter: agg[col] += dinv[row]*dinv[col] * hw[row]   (plus self-loops)
// one wave per edge, 4 features per lane
// ---------------------------------------------------------------------------
__global__ __launch_bounds__(256) void scatter_kernel(
    const int* __restrict__ ei, const float* __restrict__ dinv,
    const float* __restrict__ hw, float* __restrict__ agg)
{
    const int lane = threadIdx.x & 31;
    const long long e = (long long)blockIdx.x * 8 + (threadIdx.x >> 5);
    if (e >= (long long)(EE + NN)) return;
    int row, col;
    if (e < EE) { row = ei[e]; col = ei[EE + e]; }
    else        { row = col = (int)(e - EE); }
    const float nrm = dinv[row] * dinv[col];
    const float* __restrict__ src = hw + (size_t)row * HH;
    float* __restrict__ dst = agg + (size_t)col * HH;
#pragma unroll
    for (int t = 0; t < 4; ++t) {
        int f = lane + 32 * t;
        atomicAdd(&dst[f], nrm * src[f]);
    }
}

// ---------------------------------------------------------------------------
// BatchNorm statistics: per-feature sum and sumsq over N nodes
// ---------------------------------------------------------------------------
__global__ __launch_bounds__(256) void bn_stats_kernel(
    const float* __restrict__ agg, float* __restrict__ stats)
{
    const int f = blockIdx.x;
    const int tid = threadIdx.x;
    float s = 0.0f, ss = 0.0f;
    for (int n = tid; n < NN; n += 256) {
        float v = agg[(size_t)n * HH + f];
        s += v; ss += v * v;
    }
    __shared__ float sh0[256];
    __shared__ float sh1[256];
    sh0[tid] = s; sh1[tid] = ss;
    __syncthreads();
    for (int w = 128; w > 0; w >>= 1) {
        if (tid < w) { sh0[tid] += sh0[tid + w]; sh1[tid] += sh1[tid + w]; }
        __syncthreads();
    }
    if (tid == 0) { stats[f] = sh0[0]; stats[HH + f] = sh1[0]; }
}

__global__ void bn_params_kernel(
    const float* __restrict__ stats, const float* __restrict__ gamma,
    const float* __restrict__ beta, float* __restrict__ scale,
    float* __restrict__ shift)
{
    int f = threadIdx.x;
    if (f < HH) {
        float mu  = stats[f] * (1.0f / NN);
        float var = stats[HH + f] * (1.0f / NN) - mu * mu;
        float sc  = gamma[f] * rsqrtf(var + BN_EPS);
        scale[f] = sc;
        shift[f] = beta[f] - mu * sc;
    }
}

__global__ void bn_relu_kernel(
    const float* __restrict__ agg, const float* __restrict__ scale,
    const float* __restrict__ shift, float* __restrict__ hout)
{
    long long i = (long long)blockIdx.x * blockDim.x + threadIdx.x;
    if (i < (long long)NN * HH) {
        int f = (int)(i & (HH - 1));
        float v = fmaf(agg[i], scale[f], shift[f]);
        hout[i] = fmaxf(v, 0.0f);
    }
}

// ---------------------------------------------------------------------------
// Global mean pool + output head
// ---------------------------------------------------------------------------
__global__ void pool_zero_kernel(float* __restrict__ ssum, float* __restrict__ cnt) {
    int i = blockIdx.x * blockDim.x + threadIdx.x;
    if (i < GG * HH) ssum[i] = 0.0f;
    if (i < GG) cnt[i] = 0.0f;
}

__global__ __launch_bounds__(256) void pool_kernel(
    const float* __restrict__ h, const float* __restrict__ mask,
    const int* __restrict__ batch, float* __restrict__ ssum,
    float* __restrict__ cnt)
{
    const int lane = threadIdx.x & 31;
    const int n = blockIdx.x * 8 + (threadIdx.x >> 5);
    if (n >= NN) return;
    const int g = batch[n];
    const float mk = mask[n];
    const float* __restrict__ src = h + (size_t)n * HH;
    float* __restrict__ dst = ssum + (size_t)g * HH;
#pragma unroll
    for (int t = 0; t < 4; ++t) {
        int f = lane + 32 * t;
        atomicAdd(&dst[f], src[f] * mk);
    }
    if (lane == 0) atomicAdd(&cnt[g], 1.0f);
}

__global__ __launch_bounds__(128) void out_kernel(
    const float* __restrict__ ssum, const float* __restrict__ cnt,
    const float* __restrict__ Wout, const float* __restrict__ bout,
    float* __restrict__ out)
{
    const int g = blockIdx.x;
    const int f = threadIdx.x;
    const float c = fmaxf(cnt[g], 1.0f);
    float v = (ssum[(size_t)g * HH + f] / c) * Wout[f];
    __shared__ float sh[128];
    sh[f] = v;
    __syncthreads();
    for (int w = 64; w > 0; w >>= 1) {
        if (f < w) sh[f] += sh[f + w];
        __syncthreads();
    }
    if (f == 0) out[g] = sh[0] + bout[0];
}

// ---------------------------------------------------------------------------
// Host-side orchestration
// ---------------------------------------------------------------------------
extern "C" void kernel_launch(void* const* d_in, const int* in_sizes, int n_in,
                              void* d_out, int out_size, void* d_ws, size_t ws_size,
                              hipStream_t stream) {
    const float* x     = (const float*)d_in[0];   // [N,H]
    const float* mask  = (const float*)d_in[1];   // [N,1]
    const int*   ei    = (const int*)  d_in[2];   // [2,E]
    const int*   batch = (const int*)  d_in[3];   // [N]
    const float* W     = (const float*)d_in[4];   // [L,H,H]
    const float* b     = (const float*)d_in[5];   // [L,H]
    const float* gamma = (const float*)d_in[6];   // [L,H]
    const float* beta  = (const float*)d_in[7];   // [L,H]
    const float* Wout  = (const float*)d_in[8];   // [H,1]
    const float* bout  = (const float*)d_in[9];   // [1]
    float* out = (float*)d_out;                    // [G,1]

    // Workspace layout (floats)
    float* ws    = (float*)d_ws;
    float* hbuf  = ws;                               // N*H
    float* hw    = hbuf  + (size_t)NN * HH;          // N*H
    float* agg   = hw    + (size_t)NN * HH;          // N*H
    float* dinv  = agg   + (size_t)NN * HH;          // N
    float* stats = dinv  + NN;                       // 2*H
    float* scale = stats + 2 * HH;                   // H
    float* shift = scale + HH;                       // H
    float* ssum  = shift + HH;                       // G*H
    float* cnt   = ssum  + (size_t)GG * HH;          // G

    // 1) symmetric GCN normalization: dinv = rsqrt(deg) with self-loops
    deg_init_kernel<<<(NN + 255) / 256, 256, 0, stream>>>(dinv);
    deg_accum_kernel<<<(EE + 255) / 256, 256, 0, stream>>>(ei, dinv);
    deg_rsqrt_kernel<<<(NN + 255) / 256, 256, 0, stream>>>(dinv);

    const long long NH = (long long)NN * HH;
    const float* hin = x;
    for (int l = 0; l < LL; ++l) {
        // hw = (h * mask) @ W[l]   — WMMA
        gemm_mask_wmma_kernel<<<NN / 16, 256, 0, stream>>>(
            hin, mask, W + (size_t)l * HH * HH, hw);
        // agg = b[l]
        init_agg_kernel<<<(int)((NH + 255) / 256), 256, 0, stream>>>(b + (size_t)l * HH, agg);
        // agg += segment_sum(norm * hw[rows]) over cols (incl. self-loops)
        scatter_kernel<<<(EE + NN + 7) / 8, 256, 0, stream>>>(ei, dinv, hw, agg);
        // BatchNorm stats + fused normalize + ReLU
        bn_stats_kernel<<<HH, 256, 0, stream>>>(agg, stats);
        bn_params_kernel<<<1, HH, 0, stream>>>(stats, gamma + (size_t)l * HH,
                                               beta + (size_t)l * HH, scale, shift);
        bn_relu_kernel<<<(int)((NH + 255) / 256), 256, 0, stream>>>(agg, scale, shift, hbuf);
        hin = hbuf;
    }

    // masked global mean pool + output projection
    pool_zero_kernel<<<(GG * HH + 255) / 256, 256, 0, stream>>>(ssum, cnt);
    pool_kernel<<<(NN + 7) / 8, 256, 0, stream>>>(hbuf, mask, batch, ssum, cnt);
    out_kernel<<<GG, 128, 0, stream>>>(ssum, cnt, Wout, bout, out);
}
